// MultiViewMSRHGNN_17248588660970
// MI455X (gfx1250) — matrix operand order repaired and checked by reading, not compiled, not run on hardware
//
#include <hip/hip_runtime.h>
#include <hip/hip_bf16.h>
#include <math.h>
#include <stdint.h>

#define HID 128

typedef __attribute__((ext_vector_type(16))) __bf16 v16bf;
typedef __attribute__((ext_vector_type(8)))  __bf16 v8bf;
typedef __attribute__((ext_vector_type(8)))  float  v8f;

// pack two floats into one uint holding (bf16(lo) | bf16(hi)<<16), RNE rounding
__device__ __forceinline__ unsigned pack_bf2(float lo, float hi) {
  unsigned ul = __builtin_bit_cast(unsigned, lo);
  unsigned uh = __builtin_bit_cast(unsigned, hi);
  ul = (ul + 0x7FFFu + ((ul >> 16) & 1u)) >> 16;
  uh = (uh + 0x7FFFu + ((uh >> 16) & 1u)) & 0xFFFF0000u;
  return (ul & 0xFFFFu) | uh;
}

__device__ __forceinline__ float wred_sum(float v) {
#pragma unroll
  for (int o = 16; o > 0; o >>= 1) v += __shfl_xor(v, o, 32);
  return v;
}

// order-preserving float<->uint encoding for atomicMax on signed floats
__device__ __forceinline__ unsigned fenc(float f) {
  unsigned u = __builtin_bit_cast(unsigned, f);
  return (u >> 31) ? ~u : (u | 0x80000000u);
}
__device__ __forceinline__ float fdec(unsigned v) {
  unsigned u = (v >> 31) ? (v ^ 0x80000000u) : ~v;
  return __builtin_bit_cast(float, u);
}

// ---------------------------------------------------------------------------
// WMMA GEMM: Y[M,N] = act(X[M,K] @ W[K,N] + bias[N]).  K%32==0, N%64==0.
// Block = 128 threads (4 waves), tile 64(M)x64(N), K-step 32.
// LDS holds bf16 tiles (converted once at store time); fragments are two
// contiguous 16B ds_load_b128 reads each -- no VALU in the fragment path.
// act: 0=none 1=relu 2=gelu(exact)
// ---------------------------------------------------------------------------
__launch_bounds__(128)
__global__ void gemm_bias_act(const float* __restrict__ X, const float* __restrict__ W,
                              const float* __restrict__ bias, float* __restrict__ Y,
                              int M, int K, int N, int act) {
  // row stride 20 uints = 80 bytes (16B-aligned chunks, padded vs 64 banks)
  __shared__ unsigned sA[64][20];   // A tile: [row][kpair], k = 2*u, 2*u+1
  __shared__ unsigned sB[64][20];   // B tile transposed: [col][kpair]
  const int tid  = threadIdx.x;
  const int wave = tid >> 5;
  const int lane = tid & 31;
  const int hl   = lane >> 4;       // half-wave select
  const int m16  = lane & 15;
  const int bm = blockIdx.x * 64;
  const int bn = blockIdx.y * 64;

  v8f acc[4] = {v8f{0.f,0.f,0.f,0.f,0.f,0.f,0.f,0.f},
                v8f{0.f,0.f,0.f,0.f,0.f,0.f,0.f,0.f},
                v8f{0.f,0.f,0.f,0.f,0.f,0.f,0.f,0.f},
                v8f{0.f,0.f,0.f,0.f,0.f,0.f,0.f,0.f}};

  for (int k0 = 0; k0 < K; k0 += 32) {
    // ---- stage X tile: 64 rows x 16 k-pairs, convert f32->bf16 once ----
    for (int p = tid; p < 64 * 16; p += 128) {
      int r = p >> 4, c = p & 15;
      int row = bm + r;
      float x0 = 0.f, x1 = 0.f;
      if (row < M) {
        float2 xv = *(const float2*)(X + (size_t)row * K + k0 + 2 * c);
        x0 = xv.x; x1 = xv.y;
      }
      sA[r][c] = pack_bf2(x0, x1);
    }
    // ---- stage W tile transposed: [col][kpair] so K is contiguous ----
    for (int p = tid; p < 64 * 16; p += 128) {
      int n = p & 63, kp = p >> 6;
      float w0 = W[(size_t)(k0 + 2 * kp) * N + bn + n];
      float w1 = W[(size_t)(k0 + 2 * kp + 1) * N + bn + n];
      sB[n][kp] = pack_bf2(w0, w1);
    }
    // prefetch next K-tile into cache while we compute this one
    if (k0 + 32 < K) {
      int prow = bm + (tid >> 1);
      if (prow < M) __builtin_prefetch(X + (size_t)prow * K + k0 + 32, 0, 1);
      __builtin_prefetch(W + (size_t)(k0 + 32 + (tid & 31)) * N + bn, 0, 1);
    }
    __syncthreads();

    // A fragment: lane element e -> K = (e<8?0:16) + hl*8 + (e&7)
    // => two contiguous 8-bf16 chunks at uint offsets hl*4 and 8+hl*4
    const int arow = wave * 16 + m16;
    v8bf alo = *(const v8bf*)&sA[arow][hl * 4];
    v8bf ahi = *(const v8bf*)&sA[arow][8 + hl * 4];
    v16bf a = __builtin_shufflevector(alo, ahi, 0, 1, 2, 3, 4, 5, 6, 7,
                                      8, 9, 10, 11, 12, 13, 14, 15);
#pragma unroll
    for (int nt = 0; nt < 4; ++nt) {
      // B fragment: lane element e -> K = (e<8?0:16) + hl*8 + (e&7)
      const int brow = nt * 16 + m16;
      v8bf blo = *(const v8bf*)&sB[brow][hl * 4];
      v8bf bhi = *(const v8bf*)&sB[brow][8 + hl * 4];
      v16bf b = __builtin_shufflevector(blo, bhi, 0, 1, 2, 3, 4, 5, 6, 7,
                                        8, 9, 10, 11, 12, 13, 14, 15);
      acc[nt] = __builtin_amdgcn_wmma_f32_16x16x32_bf16(
          false, a, false, b, (short)0, acc[nt], false, false);
    }
    __syncthreads();
  }

#pragma unroll
  for (int nt = 0; nt < 4; ++nt) {
#pragma unroll
    for (int r = 0; r < 8; ++r) {
      int row = bm + wave * 16 + hl * 8 + r;   // D layout: M = 8*half + vgpr
      int col = bn + nt * 16 + m16;
      if (row < M) {
        float v = acc[nt][r] + bias[col];
        if (act == 1) v = fmaxf(v, 0.f);
        else if (act == 2) v = 0.5f * v * (1.f + erff(v * 0.70710678118654752f));
        Y[(size_t)row * N + col] = v;
      }
    }
  }
}

// ---------------------------------------------------------------------------
// utility fills
// ---------------------------------------------------------------------------
__global__ void fill_f32(float* p, float v, int n) {
  int i = blockIdx.x * 256 + threadIdx.x;
  if (i < n) p[i] = v;
}
__global__ void fill_u32(unsigned* p, unsigned v, int n) {
  int i = blockIdx.x * 256 + threadIdx.x;
  if (i < n) p[i] = v;
}

// ---------------------------------------------------------------------------
// GIN weighted-mean aggregation: wave per edge, float4 per lane (128 feats)
// ---------------------------------------------------------------------------
__launch_bounds__(256)
__global__ void edge_wmean(const float* __restrict__ x, const int* __restrict__ ei,
                           const float* __restrict__ ew, int E,
                           float* __restrict__ acc, float* __restrict__ deg) {
  int e = (blockIdx.x * 256 + threadIdx.x) >> 5;
  int lane = threadIdx.x & 31;
  if (e >= E) return;
  int s = ei[e], d = ei[E + e];
  float w = ew[e];
  float4 v = ((const float4*)(x + (size_t)s * HID))[lane];
  float* o = acc + (size_t)d * HID + lane * 4;
  atomicAdd(o + 0, v.x * w); atomicAdd(o + 1, v.y * w);
  atomicAdd(o + 2, v.z * w); atomicAdd(o + 3, v.w * w);
  if (lane == 0) atomicAdd(&deg[d], w);
}

__global__ void gin_combine_k(const float* __restrict__ h, const float* __restrict__ acc,
                              const float* __restrict__ deg, const float* __restrict__ eps,
                              float* __restrict__ out, int total) {
  int i = blockIdx.x * 256 + threadIdx.x;
  if (i >= total) return;
  int n = i >> 7;
  out[i] = (1.f + eps[0]) * h[i] + acc[i] / fmaxf(deg[n], 1e-8f);
}

// ---------------------------------------------------------------------------
// multi-view pooling (mean/max/sum over nodes) + scorer + softmax + combine
// ---------------------------------------------------------------------------
__launch_bounds__(256)
__global__ void pool_views(const float* __restrict__ r0, const float* __restrict__ r1,
                           const float* __restrict__ r2, int N, float* __restrict__ pooled) {
  __shared__ float ssum[256], smax[256];
  int r = blockIdx.x >> 7, f = blockIdx.x & 127;
  const float* rep = (r == 0) ? r0 : ((r == 1) ? r1 : r2);
  float s = 0.f, mx = -INFINITY;
  for (int i = threadIdx.x; i < N; i += 256) {
    float v = rep[(size_t)i * HID + f];
    s += v; mx = fmaxf(mx, v);
  }
  ssum[threadIdx.x] = s; smax[threadIdx.x] = mx;
  __syncthreads();
  for (int o = 128; o > 0; o >>= 1) {
    if (threadIdx.x < o) {
      ssum[threadIdx.x] += ssum[threadIdx.x + o];
      smax[threadIdx.x] = fmaxf(smax[threadIdx.x], smax[threadIdx.x + o]);
    }
    __syncthreads();
  }
  if (threadIdx.x == 0) {
    pooled[r * 384 + f]       = ssum[0] / (float)N;
    pooled[r * 384 + 128 + f] = smax[0];
    pooled[r * 384 + 256 + f] = ssum[0];
  }
}

__launch_bounds__(128)
__global__ void fuse_score(const float* __restrict__ pooled, const float* __restrict__ w1,
                           const float* __restrict__ b1, const float* __restrict__ w2,
                           const float* __restrict__ b2, float* __restrict__ wview) {
  __shared__ float red[128];
  __shared__ float sc[3];
  int t = threadIdx.x;
  for (int r = 0; r < 3; ++r) {
    float h = b1[t];
    for (int j = 0; j < 384; ++j) h += pooled[r * 384 + j] * w1[j * 128 + t];
    red[t] = tanhf(h) * w2[t];
    __syncthreads();
    for (int o = 64; o > 0; o >>= 1) {
      if (t < o) red[t] += red[t + o];
      __syncthreads();
    }
    if (t == 0) sc[r] = red[0] + b2[0];
    __syncthreads();
  }
  if (t == 0) {
    float m = fmaxf(sc[0], fmaxf(sc[1], sc[2]));
    float e0 = expf(sc[0] - m), e1 = expf(sc[1] - m), e2 = expf(sc[2] - m);
    float s = e0 + e1 + e2;
    wview[0] = e0 / s; wview[1] = e1 / s; wview[2] = e2 / s;
  }
}

__global__ void fuse_combine_k(const float* __restrict__ r0, const float* __restrict__ r1,
                               const float* __restrict__ r2, const float* __restrict__ wv,
                               float* __restrict__ out, int total) {
  int i = blockIdx.x * 256 + threadIdx.x;
  if (i >= total) return;
  out[i] = wv[0] * r0[i] + wv[1] * r1[i] + wv[2] * r2[i];
}

// ---------------------------------------------------------------------------
// hash-based edge merge (dedup + weight-sum), replacing reference's sort
// ---------------------------------------------------------------------------
__global__ void hash_insert(const int* __restrict__ ei, const float* __restrict__ ew,
                            const float* __restrict__ wview, int view, int E, int n,
                            unsigned* __restrict__ keys, float* __restrict__ vals,
                            int* __restrict__ owner, int base, unsigned mask) {
  int e = blockIdx.x * 256 + threadIdx.x;
  if (e >= E) return;
  unsigned key = (unsigned)ei[e] * (unsigned)n + (unsigned)ei[E + e];
  float w = ew[e] * wview[view];
  unsigned h = (key * 2654435761u) & mask;
  for (;;) {
    unsigned prev = atomicCAS(&keys[h], 0xFFFFFFFFu, key);
    if (prev == 0xFFFFFFFFu || prev == key) {
      atomicAdd(&vals[h], w);
      if (prev == 0xFFFFFFFFu) owner[h] = base + e;   // CAS winner is representative
      break;
    }
    h = (h + 1) & mask;
  }
}

__global__ void hash_finalize(const int* __restrict__ ei, int E, int n, int base,
                              const unsigned* __restrict__ keys, const float* __restrict__ vals,
                              const int* __restrict__ owner, unsigned mask,
                              float* __restrict__ mw, int* __restrict__ valid) {
  int e = blockIdx.x * 256 + threadIdx.x;
  if (e >= E) return;
  unsigned key = (unsigned)ei[e] * (unsigned)n + (unsigned)ei[E + e];
  unsigned h = (key * 2654435761u) & mask;
  while (keys[h] != key) h = (h + 1) & mask;
  int id = base + e;
  int own = (owner[h] == id);
  valid[id] = own;
  mw[id] = own ? vals[h] : 0.f;
}

__global__ void copy_edges(const int* __restrict__ ei, int E, int base,
                           int* __restrict__ msrc, int* __restrict__ mdst) {
  int e = blockIdx.x * 256 + threadIdx.x;
  if (e >= E) return;
  msrc[base + e] = ei[e];
  mdst[base + e] = ei[E + e];
}

__global__ void passthrough_edges(const int* __restrict__ ei, const float* __restrict__ ew,
                                  int E, int* __restrict__ msrc, int* __restrict__ mdst,
                                  float* __restrict__ mw, int* __restrict__ valid) {
  int e = blockIdx.x * 256 + threadIdx.x;
  if (e >= E) return;
  msrc[e] = ei[e]; mdst[e] = ei[E + e]; mw[e] = ew[e]; valid[e] = 1;
}

// ---------------------------------------------------------------------------
// transformer edge attention
// ---------------------------------------------------------------------------
__launch_bounds__(256)
__global__ void edge_scores(const float* __restrict__ q, const float* __restrict__ k,
                            const int* __restrict__ src, const int* __restrict__ dst,
                            const float* __restrict__ mw, const int* __restrict__ valid,
                            const float* __restrict__ edge_w, const float* __restrict__ edge_b,
                            int E, float* __restrict__ score, unsigned* __restrict__ mxb) {
  int e = (blockIdx.x * 256 + threadIdx.x) >> 5;
  int lane = threadIdx.x & 31;
  if (e >= E) return;
  int s = src[e], d = dst[e];
  float4 qv = ((const float4*)(q + (size_t)d * HID))[lane];
  float4 kv = ((const float4*)(k + (size_t)s * HID))[lane];
  float p = qv.x * kv.x + qv.y * kv.y + qv.z * kv.z + qv.w * kv.w;
  p = wred_sum(p);
  if (lane == 0) {
    float sc = valid[e] ? (p * 0.08838834764831843f + mw[e] * edge_w[0] + edge_b[0])
                        : -INFINITY;
    score[e] = sc;
    atomicMax(&mxb[d], fenc(sc));
  }
}

__global__ void edge_exp(const int* __restrict__ dst, int E, const unsigned* __restrict__ mxb,
                         float* __restrict__ score_eo, float* __restrict__ denom) {
  int e = blockIdx.x * 256 + threadIdx.x;
  if (e >= E) return;
  int d = dst[e];
  float m = fdec(mxb[d]);
  if (!__builtin_isfinite(m)) m = 0.f;
  float ee = expf(score_eo[e] - m);
  score_eo[e] = ee;
  atomicAdd(&denom[d], ee);
}

__launch_bounds__(256)
__global__ void edge_agg(const float* __restrict__ v, const int* __restrict__ src,
                         const int* __restrict__ dst, const float* __restrict__ earr,
                         const float* __restrict__ denom, int E, float* __restrict__ agg) {
  int e = (blockIdx.x * 256 + threadIdx.x) >> 5;
  int lane = threadIdx.x & 31;
  if (e >= E) return;
  int s = src[e], d = dst[e];
  float a = earr[e] / fmaxf(denom[d], 1e-12f);
  float4 vv = ((const float4*)(v + (size_t)s * HID))[lane];
  float* o = agg + (size_t)d * HID + lane * 4;
  atomicAdd(o + 0, vv.x * a); atomicAdd(o + 1, vv.y * a);
  atomicAdd(o + 2, vv.z * a); atomicAdd(o + 3, vv.w * a);
}

// residual add (2 or 3 way) + LayerNorm over 128 feats; wave per node
__launch_bounds__(256)
__global__ void residual_ln(const float* __restrict__ x, const float* __restrict__ a,
                            const float* __restrict__ c, int has_c,
                            const float* __restrict__ g, const float* __restrict__ beta,
                            float* __restrict__ y, int N) {
  int w = (blockIdx.x * 256 + threadIdx.x) >> 5;
  int lane = threadIdx.x & 31;
  if (w >= N) return;
  float4 t = ((const float4*)(x + (size_t)w * HID))[lane];
  float4 av = ((const float4*)(a + (size_t)w * HID))[lane];
  t.x += av.x; t.y += av.y; t.z += av.z; t.w += av.w;
  if (has_c) {
    float4 cv = ((const float4*)(c + (size_t)w * HID))[lane];
    t.x += cv.x; t.y += cv.y; t.z += cv.z; t.w += cv.w;
  }
  float mean = wred_sum(t.x + t.y + t.z + t.w) * (1.0f / HID);
  float dx = t.x - mean, dy = t.y - mean, dz = t.z - mean, dw = t.w - mean;
  float var = wred_sum(dx * dx + dy * dy + dz * dz + dw * dw) * (1.0f / HID);
  float rstd = rsqrtf(var + 1e-5f);
  float4 gv = ((const float4*)g)[lane];
  float4 bv = ((const float4*)beta)[lane];
  float4 o;
  o.x = dx * rstd * gv.x + bv.x; o.y = dy * rstd * gv.y + bv.y;
  o.z = dz * rstd * gv.z + bv.z; o.w = dw * rstd * gv.w + bv.w;
  ((float4*)(y + (size_t)w * HID))[lane] = o;
}

// ---------------------------------------------------------------------------
// host-side orchestration
// ---------------------------------------------------------------------------
static void gemm(hipStream_t st, const float* X, const float* W, const float* B,
                 float* Y, int M, int K, int N, int act) {
  dim3 g((unsigned)((M + 63) / 64), (unsigned)(N / 64));
  gemm_bias_act<<<g, dim3(128), 0, st>>>(X, W, B, Y, M, K, N, act);
}
static void fillf(hipStream_t st, float* p, float v, size_t n) {
  fill_f32<<<(unsigned)((n + 255) / 256), 256, 0, st>>>(p, v, (int)n);
}
static void fillu(hipStream_t st, unsigned* p, unsigned v, size_t n) {
  fill_u32<<<(unsigned)((n + 255) / 256), 256, 0, st>>>(p, v, (int)n);
}

// tr leaves at base: bn1,bn2,edge_b,edge_w,fb1,fb2,fw1,fw2,g1,g2,kb,kw,qb,qw,sb,sw,vb,vw
static void run_transformer(hipStream_t st, const float* xin, int N,
                            const int* msrc, const int* mdst, const float* mw,
                            const int* valid, int E, void* const* tr,
                            float* q, float* k, float* v, float* s, float* agg,
                            float* x1, float* mid, float* ffo,
                            float* score, unsigned* mxb, float* denom, float* out) {
  const float* bn1 = (const float*)tr[0];  const float* bn2 = (const float*)tr[1];
  const float* edge_b = (const float*)tr[2]; const float* edge_w = (const float*)tr[3];
  const float* fb1 = (const float*)tr[4];  const float* fb2 = (const float*)tr[5];
  const float* fw1 = (const float*)tr[6];  const float* fw2 = (const float*)tr[7];
  const float* g1 = (const float*)tr[8];   const float* g2 = (const float*)tr[9];
  const float* kb = (const float*)tr[10];  const float* kw = (const float*)tr[11];
  const float* qb = (const float*)tr[12];  const float* qw = (const float*)tr[13];
  const float* sb = (const float*)tr[14];  const float* sw = (const float*)tr[15];
  const float* vb = (const float*)tr[16];  const float* vw = (const float*)tr[17];

  gemm(st, xin, qw, qb, q, N, HID, HID, 0);
  gemm(st, xin, kw, kb, k, N, HID, HID, 0);
  gemm(st, xin, vw, vb, v, N, HID, HID, 0);
  gemm(st, xin, sw, sb, s, N, HID, HID, 0);

  fillu(st, mxb, 0x007FFFFFu /* enc(-inf) */, N);
  fillf(st, denom, 0.f, N);
  fillf(st, agg, 0.f, (size_t)N * HID);

  unsigned ewb = (unsigned)((E + 7) / 8);
  edge_scores<<<ewb, 256, 0, st>>>(q, k, msrc, mdst, mw, valid, edge_w, edge_b, E, score, mxb);
  edge_exp<<<(unsigned)((E + 255) / 256), 256, 0, st>>>(mdst, E, mxb, score, denom);
  edge_agg<<<ewb, 256, 0, st>>>(v, msrc, mdst, score, denom, E, agg);

  unsigned nwb = (unsigned)((N + 7) / 8);
  residual_ln<<<nwb, 256, 0, st>>>(xin, agg, s, 1, g1, bn1, x1, N);
  gemm(st, x1, fw1, fb1, mid, N, HID, 2 * HID, 2);
  gemm(st, mid, fw2, fb2, ffo, N, 2 * HID, HID, 0);
  residual_ln<<<nwb, 256, 0, st>>>(x1, ffo, nullptr, 0, g2, bn2, out, N);
}

struct Arena {
  char* cur;
  explicit Arena(void* p) : cur((char*)p) {}
  void* get(size_t bytes) {
    uintptr_t p = ((uintptr_t)cur + 255) & ~(uintptr_t)255;
    cur = (char*)p + bytes;
    return (void*)p;
  }
};

static void run_gnn_branch(hipStream_t st, void* ws, const float* X, int N,
                           const int* ei0, const float* ew0, const int* ei1, const float* ew1,
                           const int* ei2, const float* ew2, int E,
                           const float* projw, const float* projb,
                           void* const* g0, void* const* g1p, void* const* g2p,
                           void* const* fus, void* const* tr, float* out) {
  Arena ar(ws);
  size_t NH = (size_t)N * HID;
  int Etot = 3 * E;
  float* h     = (float*)ar.get(NH * 4);
  float* rep0  = (float*)ar.get(NH * 4);
  float* rep1  = (float*)ar.get(NH * 4);
  float* rep2  = (float*)ar.get(NH * 4);
  float* acc   = (float*)ar.get(NH * 4);            // wmean acc, reused as agg
  float* gh    = (float*)ar.get(NH * 4);            // gin input, reused as ffn out
  float* mid   = (float*)ar.get((size_t)N * 256 * 4);
  float* fused = (float*)ar.get(NH * 4);
  float* q     = (float*)ar.get(NH * 4);
  float* kk    = (float*)ar.get(NH * 4);
  float* vv    = (float*)ar.get(NH * 4);
  float* ss    = (float*)ar.get(NH * 4);
  float* x1    = (float*)ar.get(NH * 4);
  float* deg   = (float*)ar.get((size_t)N * 4);
  float* denom = (float*)ar.get((size_t)N * 4);
  unsigned* mxb = (unsigned*)ar.get((size_t)N * 4);
  float* score = (float*)ar.get((size_t)Etot * 4);
  float* mw    = (float*)ar.get((size_t)Etot * 4);
  int* valid   = (int*)ar.get((size_t)Etot * 4);
  int* msrc    = (int*)ar.get((size_t)Etot * 4);
  int* mdst    = (int*)ar.get((size_t)Etot * 4);
  float* pooled = (float*)ar.get(3 * 384 * 4);
  float* wview  = (float*)ar.get(64);
  const int HS = 1 << 21;
  unsigned* hkeys = (unsigned*)ar.get((size_t)HS * 4);
  float* hvals    = (float*)ar.get((size_t)HS * 4);
  int* howner     = (int*)ar.get((size_t)HS * 4);

  const int* eis[3] = {ei0, ei1, ei2};
  const float* ews[3] = {ew0, ew1, ew2};
  void* const* gins[3] = {g0, g1p, g2p};
  float* reps[3] = {rep0, rep1, rep2};

  gemm(st, X, projw, projb, h, N, 256, HID, 0);

  for (int v = 0; v < 3; ++v) {
    fillf(st, acc, 0.f, NH);
    fillf(st, deg, 0.f, (size_t)N);
    unsigned eb = (unsigned)((E + 7) / 8);
    edge_wmean<<<eb, 256, 0, st>>>(h, eis[v], ews[v], E, acc, deg);
    unsigned tb = (unsigned)((NH + 255) / 256);
    // gin leaves: b1,b2,eps,w1,w2
    gin_combine_k<<<tb, 256, 0, st>>>(h, acc, deg, (const float*)gins[v][2], gh, (int)NH);
    gemm(st, gh,  (const float*)gins[v][3], (const float*)gins[v][0], mid,     N, HID, HID, 1);
    gemm(st, mid, (const float*)gins[v][4], (const float*)gins[v][1], reps[v], N, HID, HID, 0);
  }

  pool_views<<<3 * HID, 256, 0, st>>>(rep0, rep1, rep2, N, pooled);
  // fus leaves: b1,b2,w1,w2
  fuse_score<<<1, 128, 0, st>>>(pooled, (const float*)fus[2], (const float*)fus[0],
                                (const float*)fus[3], (const float*)fus[1], wview);
  fuse_combine_k<<<(unsigned)((NH + 255) / 256), 256, 0, st>>>(rep0, rep1, rep2, wview,
                                                               fused, (int)NH);

  fillu(st, hkeys, 0xFFFFFFFFu, (size_t)HS);
  fillf(st, hvals, 0.f, (size_t)HS);
  for (int v = 0; v < 3; ++v) {
    unsigned eb = (unsigned)((E + 255) / 256);
    hash_insert<<<eb, 256, 0, st>>>(eis[v], ews[v], wview, v, E, N, hkeys, hvals,
                                    howner, v * E, (unsigned)(HS - 1));
    copy_edges<<<eb, 256, 0, st>>>(eis[v], E, v * E, msrc, mdst);
  }
  for (int v = 0; v < 3; ++v) {
    unsigned eb = (unsigned)((E + 255) / 256);
    hash_finalize<<<eb, 256, 0, st>>>(eis[v], E, N, v * E, hkeys, hvals, howner,
                                      (unsigned)(HS - 1), mw, valid);
  }

  run_transformer(st, fused, N, msrc, mdst, mw, valid, Etot, tr,
                  q, kk, vv, ss, acc /*agg*/, x1, mid, gh /*ffo*/,
                  score, mxb, denom, out);
}

static void run_gene_branch(hipStream_t st, void* ws, const float* X, int N,
                            const int* ei, const float* ew, int E,
                            const float* projw, const float* projb,
                            void* const* tr, float* out) {
  Arena ar(ws);
  size_t NH = (size_t)N * HID;
  float* h   = (float*)ar.get(NH * 4);
  float* q   = (float*)ar.get(NH * 4);
  float* kk  = (float*)ar.get(NH * 4);
  float* vv  = (float*)ar.get(NH * 4);
  float* ss  = (float*)ar.get(NH * 4);
  float* agg = (float*)ar.get(NH * 4);
  float* x1  = (float*)ar.get(NH * 4);
  float* ffo = (float*)ar.get(NH * 4);
  float* mid = (float*)ar.get((size_t)N * 256 * 4);
  float* denom = (float*)ar.get((size_t)N * 4);
  unsigned* mxb = (unsigned*)ar.get((size_t)N * 4);
  float* score = (float*)ar.get((size_t)E * 4);
  float* mw    = (float*)ar.get((size_t)E * 4);
  int* valid   = (int*)ar.get((size_t)E * 4);
  int* msrc    = (int*)ar.get((size_t)E * 4);
  int* mdst    = (int*)ar.get((size_t)E * 4);

  gemm(st, X, projw, projb, h, N, 256, HID, 0);
  passthrough_edges<<<(unsigned)((E + 255) / 256), 256, 0, st>>>(ei, ew, E, msrc, mdst,
                                                                 mw, valid);
  run_transformer(st, h, N, msrc, mdst, mw, valid, E, tr,
                  q, kk, vv, ss, agg, x1, mid, ffo, score, mxb, denom, out);
}

extern "C" void kernel_launch(void* const* d_in, const int* in_sizes, int n_in,
                              void* d_out, int out_size, void* d_ws, size_t ws_size,
                              hipStream_t stream) {
  (void)n_in; (void)out_size; (void)ws_size;
  auto F = [&](int i) { return (const float*)d_in[i]; };
  auto I = [&](int i) { return (const int*)d_in[i]; };

  const int RAWD = 256;
  const int ND = in_sizes[0] / RAWD;
  const int NS = in_sizes[1] / RAWD;
  const int NG = in_sizes[2] / RAWD;
  const int ED = in_sizes[4];
  const int ES = in_sizes[10];
  const int EG = in_sizes[16];

  float* out_drug = (float*)d_out;
  float* out_dis  = out_drug + (size_t)ND * HID;
  float* out_gene = out_dis + (size_t)NS * HID;

  // param leaf indices assuming jax pytree flatten (sorted dict keys):
  // 17 dis_fus(4), 21 dis_gin(3x5), 36 dis_proj(b,w), 38 dis_tr(18),
  // 56 drug_fus, 60 drug_gin, 75 drug_proj(b,w), 77 drug_tr,
  // 95 gene_proj(b,w), 97 gene_tr
  run_gnn_branch(stream, d_ws, F(0), ND,
                 I(3), F(4), I(5), F(6), I(7), F(8), ED,
                 F(76), F(75),
                 d_in + 60, d_in + 65, d_in + 70,
                 d_in + 56, d_in + 77, out_drug);

  run_gnn_branch(stream, d_ws, F(1), NS,
                 I(9), F(10), I(11), F(12), I(13), F(14), ES,
                 F(37), F(36),
                 d_in + 21, d_in + 26, d_in + 31,
                 d_in + 17, d_in + 38, out_dis);

  run_gene_branch(stream, d_ws, F(2), NG, I(15), F(16), EG,
                  F(96), F(95), d_in + 97, out_gene);
}
